// CausalSelfAttention_29824252903618
// MI455X (gfx1250) — matrix operand order repaired and checked by reading, not compiled, not run on hardware
//
#include <hip/hip_runtime.h>
#include <hip/hip_bf16.h>

// ---------------------------------------------------------------------------
// CausalSelfAttention forward for gfx1250 (MI455X), wave32 + WMMA.
// Pipeline: QKV GEMM (wmma f16, double-buffered LDS) -> RoPE(+K/V transpose to
// outputs) -> flash attention (wmma f16, online softmax) -> proj GEMM (wmma).
// Workspace layout (floats): [0, 4096*3072) = qkv ; then 4096*2048 = y_attn.
// d_out layout (floats): y (4*1024*2048) | k (4*4*1024*128) | v (4*4*1024*128)
// ---------------------------------------------------------------------------

#define B_ 4
#define T_ 1024
#define C_ 2048
#define NH_ 16
#define NKV_ 4
#define HD_ 128
#define QKVN_ 3072   // C + 2*NKV*HD

typedef float     v8f  __attribute__((ext_vector_type(8)));
typedef _Float16  v8h  __attribute__((ext_vector_type(8)));
typedef _Float16  v16h __attribute__((ext_vector_type(16)));

__device__ __forceinline__ v8f zero8() {
    v8f z;
#pragma unroll
    for (int i = 0; i < 8; ++i) z[i] = 0.0f;
    return z;
}

// A-matrix fragment (16x32 f16): lane half 0 -> K {0..7, 16..23},
// lane half 1 -> K {8..15, 24..31}; row = lane&15 (caller passes row pointer).
__device__ __forceinline__ v16h fragA(const _Float16* rowp, int lhalf) {
    const int kb = lhalf * 8;
    v8h a0 = *(const v8h*)(rowp + kb);
    v8h a1 = *(const v8h*)(rowp + kb + 16);
    v16h r;
#pragma unroll
    for (int i = 0; i < 8; ++i) { r[i] = a0[i]; r[i + 8] = a1[i]; }
    return r;
}

// B-matrix fragment (32x16 f16): lane half 0 -> K {0..7, 16..23}, half 1 -> K {8..15, 24..31},
// column n = lane&15 (caller passes pointer to that column's K-contiguous row).
__device__ __forceinline__ v16h fragB(const _Float16* colp, int lhalf) {
    const int kb = lhalf * 8;
    v8h b0 = *(const v8h*)(colp + kb);
    v8h b1 = *(const v8h*)(colp + kb + 16);
    v16h r;
#pragma unroll
    for (int i = 0; i < 8; ++i) { r[i] = b0[i]; r[i + 8] = b1[i]; }
    return r;
}

__device__ __forceinline__ v8f wmma_f16(v16h a, v16h b, v8f c) {
    return __builtin_amdgcn_wmma_f32_16x16x32_f16(false, a, false, b,
                                                  (short)0, c, false, false);
}

// ---------------------------------------------------------------------------
// Out[M][N] = A[M][K] @ W[N][K]^T + bias[N]  (fp32 memory, f16 WMMA math).
// Block tile 64x128, 8 waves (2x4), wave tile 32x32, BK=64, double-buffered
// LDS so global loads of tile t+1 overlap the 8 WMMAs of tile t.
// All fragments are preloaded into distinct registers before the WMMA burst
// so ds_load_b128s clause together and WMMAs issue back-to-back.
// ---------------------------------------------------------------------------
template <int N, int K>
__global__ __launch_bounds__(256)
void gemm_bias_kernel(const float* __restrict__ A, const float* __restrict__ W,
                      const float* __restrict__ bias, float* __restrict__ Out) {
    __shared__ alignas(16) _Float16 Ash[2][64][72];
    __shared__ alignas(16) _Float16 Bsh[2][128][72];

    const int tid   = threadIdx.x;
    const int lane  = tid & 31;
    const int wid   = tid >> 5;
    const int lmod  = lane & 15;
    const int lhalf = lane >> 4;

    const int m0 = blockIdx.y * 64;
    const int n0 = blockIdx.x * 128;
    const int wm = (wid >> 2) * 32;   // 0 or 32
    const int wn = (wid & 3) * 32;    // 0,32,64,96

    const int arow = tid >> 2;        // 0..63
    const int akk  = (tid & 3) * 16;  // 0,16,32,48

    v8f acc[2][2];
#pragma unroll
    for (int mt = 0; mt < 2; ++mt)
#pragma unroll
        for (int nt = 0; nt < 2; ++nt) acc[mt][nt] = zero8();

    // stage tile at k0 into LDS buffer `buf` (fp32 -> f16)
    auto stage = [&](int buf, int k0) {
        {
            const float* src = A + (size_t)(m0 + arow) * K + k0 + akk;
#pragma unroll
            for (int j = 0; j < 16; ++j) Ash[buf][arow][akk + j] = (_Float16)src[j];
        }
#pragma unroll
        for (int it = 0; it < 2; ++it) {
            const int brow = arow + it * 64;
            const float* src = W + (size_t)(n0 + brow) * K + k0 + akk;
#pragma unroll
            for (int j = 0; j < 16; ++j) Bsh[buf][brow][akk + j] = (_Float16)src[j];
        }
    };

    stage(0, 0);
    __syncthreads();

    const int nkt = K / 64;
    for (int kt = 0; kt < nkt; ++kt) {
        const int cur = kt & 1;
        if (kt + 1 < nkt) {
            stage(cur ^ 1, (kt + 1) * 64);
            __builtin_prefetch(A + (size_t)(m0 + arow) * K + (kt + 2) * 64 + akk, 0, 1);
            __builtin_prefetch(W + (size_t)(n0 + arow) * K + (kt + 2) * 64 + akk, 0, 1);
        }
        // preload all 8 fragments, then fire all 8 WMMAs
        v16h af[2][2], bf[2][2];
#pragma unroll
        for (int sub = 0; sub < 2; ++sub) {
#pragma unroll
            for (int mt = 0; mt < 2; ++mt)
                af[sub][mt] = fragA(&Ash[cur][wm + mt * 16 + lmod][sub * 32], lhalf);
#pragma unroll
            for (int nt = 0; nt < 2; ++nt)
                bf[sub][nt] = fragB(&Bsh[cur][wn + nt * 16 + lmod][sub * 32], lhalf);
        }
#pragma unroll
        for (int sub = 0; sub < 2; ++sub)
#pragma unroll
            for (int mt = 0; mt < 2; ++mt)
#pragma unroll
                for (int nt = 0; nt < 2; ++nt)
                    acc[mt][nt] = wmma_f16(af[sub][mt], bf[sub][nt], acc[mt][nt]);
        __syncthreads();
    }

    // bias + store (D layout: row = r + lhalf*8, col = lmod within 16-tile)
#pragma unroll
    for (int mt = 0; mt < 2; ++mt) {
#pragma unroll
        for (int nt = 0; nt < 2; ++nt) {
            const int col = n0 + wn + nt * 16 + lmod;
            const float bv = bias[col];
            float* op = Out + (size_t)(m0 + wm + mt * 16 + lhalf * 8) * N + col;
#pragma unroll
            for (int r = 0; r < 8; ++r)
                op[(size_t)r * N] = acc[mt][nt][r] + bv;
        }
    }
}

// ---------------------------------------------------------------------------
// RoPE + transpose/copy of K,V into output buffers.
// role 0: rope Q in place inside qkv workspace
// role 1: rope K from qkv -> outK[(b,kv,t,hd)]
// role 2: copy V from qkv -> outV[(b,kv,t,hd)]
// ---------------------------------------------------------------------------
__global__ __launch_bounds__(256)
void rope_copy_kernel(float* __restrict__ qkv, float* __restrict__ outK,
                      float* __restrict__ outV, const float* __restrict__ fcos,
                      const float* __restrict__ fsin, int role, int total) {
    const int idx = blockIdx.x * blockDim.x + threadIdx.x;
    if (idx >= total) return;

    if (role == 0) {
        const int i = idx & 63;
        int r = idx >> 6;
        const int head = r & 15; r >>= 4;
        const int t = r & 1023;
        const int b = r >> 10;
        float* p = qkv + ((size_t)(b * T_ + t)) * QKVN_ + head * HD_ + 2 * i;
        const float c = fcos[t * 64 + i], s = fsin[t * 64 + i];
        const float re = p[0], im = p[1];
        p[0] = re * c - im * s;
        p[1] = re * s + im * c;
    } else if (role == 1) {
        const int i = idx & 63;
        int r = idx >> 6;
        const int kvh = r & 3; r >>= 2;
        const int t = r & 1023;
        const int b = r >> 10;
        const float* p = qkv + ((size_t)(b * T_ + t)) * QKVN_ + C_ + kvh * HD_ + 2 * i;
        const float c = fcos[t * 64 + i], s = fsin[t * 64 + i];
        const float re = p[0], im = p[1];
        float* q = outK + (((size_t)(b * NKV_ + kvh)) * T_ + t) * HD_ + 2 * i;
        q[0] = re * c - im * s;
        q[1] = re * s + im * c;
    } else {
        const int d = idx & 127;
        int r = idx >> 7;
        const int kvh = r & 3; r >>= 2;
        const int t = r & 1023;
        const int b = r >> 10;
        outV[(((size_t)(b * NKV_ + kvh)) * T_ + t) * HD_ + d] =
            qkv[((size_t)(b * T_ + t)) * QKVN_ + C_ + NKV_ * HD_ + kvh * HD_ + d];
    }
}

// ---------------------------------------------------------------------------
// Flash attention: block = (qtile of 64 rows, head, batch), 4 waves,
// each wave owns 16 q rows. Iterates s in tiles of 32 with online softmax.
// ---------------------------------------------------------------------------
__global__ __launch_bounds__(128)
void attn_kernel(const float* __restrict__ qkv, const float* __restrict__ Kg,
                 const float* __restrict__ Vg, float* __restrict__ Y) {
    __shared__ alignas(16) _Float16 Ksh[32][136];      // [s][hd], padded
    __shared__ alignas(16) _Float16 VshT[128][40];     // [hd][s], padded
    __shared__ alignas(16) _Float16 Psh[4][16][40];    // per-wave P transpose

    const int tid   = threadIdx.x;
    const int lane  = tid & 31;
    const int w     = tid >> 5;
    const int lmod  = lane & 15;
    const int lhalf = lane >> 4;

    const int h     = blockIdx.y;
    const int b     = blockIdx.z;
    const int g     = h >> 2;             // kv head (rep = 4)
    const int qbase = blockIdx.x * 64;
    const float scale = 0.08838834764831845f;   // 1/sqrt(128)

    // Q fragments: 16 rows x 128 (4 K-chunks of 32), fp32 -> f16
    v16h qf[4];
    {
        const int qrow = qbase + w * 16 + lmod;
        const float* qp = qkv + (size_t)(b * T_ + qrow) * QKVN_ + h * HD_;
#pragma unroll
        for (int kc = 0; kc < 4; ++kc) {
            const int kb = kc * 32 + lhalf * 8;
            v16h r;
#pragma unroll
            for (int i = 0; i < 8; ++i) {
                r[i]     = (_Float16)qp[kb + i];
                r[i + 8] = (_Float16)qp[kb + 16 + i];
            }
            qf[kc] = r;
        }
    }

    v8f o[8];
#pragma unroll
    for (int nt = 0; nt < 8; ++nt) o[nt] = zero8();
    float mi[8], li[8];
#pragma unroll
    for (int r = 0; r < 8; ++r) { mi[r] = -1e30f; li[r] = 0.0f; }

    const int nst  = (qbase + 64) >> 5;   // number of 32-wide s tiles
    const int srow = tid >> 2;            // 0..31
    const int hdc  = (tid & 3) * 32;

    const float* Kbase = Kg + ((size_t)(b * NKV_ + g) * T_) * HD_;
    const float* Vbase = Vg + ((size_t)(b * NKV_ + g) * T_) * HD_;

    for (int st = 0; st < nst; ++st) {
        const int s0 = st * 32;
        // cooperative staging: K row-major, V transposed, fp32 -> f16
        {
            const float* ks = Kbase + (size_t)(s0 + srow) * HD_ + hdc;
            const float* vs = Vbase + (size_t)(s0 + srow) * HD_ + hdc;
#pragma unroll
            for (int j = 0; j < 8; ++j) {
                const float4 kf4 = ((const float4*)ks)[j];
                const float4 vf4 = ((const float4*)vs)[j];
                const int hh = hdc + j * 4;
                Ksh[srow][hh + 0] = (_Float16)kf4.x;
                Ksh[srow][hh + 1] = (_Float16)kf4.y;
                Ksh[srow][hh + 2] = (_Float16)kf4.z;
                Ksh[srow][hh + 3] = (_Float16)kf4.w;
                VshT[hh + 0][srow] = (_Float16)vf4.x;
                VshT[hh + 1][srow] = (_Float16)vf4.y;
                VshT[hh + 2][srow] = (_Float16)vf4.z;
                VshT[hh + 3][srow] = (_Float16)vf4.w;
            }
        }
        // prefetch next tile while this one is consumed
        if (st + 1 < nst) {
            __builtin_prefetch(Kbase + (size_t)(s0 + 32 + srow) * HD_ + hdc, 0, 1);
            __builtin_prefetch(Vbase + (size_t)(s0 + 32 + srow) * HD_ + hdc, 0, 1);
        }
        __syncthreads();

        // preload all K fragments (distinct regs), then the 8 QK^T WMMAs
        v16h kf[4][2];
#pragma unroll
        for (int kc = 0; kc < 4; ++kc) {
            kf[kc][0] = fragB(&Ksh[lmod][kc * 32], lhalf);
            kf[kc][1] = fragB(&Ksh[16 + lmod][kc * 32], lhalf);
        }
        v8f sa = zero8(), sb = zero8();
#pragma unroll
        for (int kc = 0; kc < 4; ++kc) {
            sa = wmma_f16(qf[kc], kf[kc][0], sa);
            sb = wmma_f16(qf[kc], kf[kc][1], sb);
        }

        // online softmax (per row r; row = qbase + w*16 + r + lhalf*8)
#pragma unroll
        for (int r = 0; r < 8; ++r) {
            const int row = qbase + w * 16 + r + lhalf * 8;
            float v0 = sa[r] * scale;
            float v1 = sb[r] * scale;
            if (s0 + lmod > row)      v0 = -1e30f;
            if (s0 + 16 + lmod > row) v1 = -1e30f;
            float tmax = fmaxf(v0, v1);
#pragma unroll
            for (int off = 1; off < 16; off <<= 1)
                tmax = fmaxf(tmax, __shfl_xor(tmax, off, 32));
            const float mnew  = fmaxf(mi[r], tmax);
            const float alpha = __expf(mi[r] - mnew);
            mi[r] = mnew;
            const float p0 = __expf(v0 - mnew);
            const float p1 = __expf(v1 - mnew);
            float rsum = p0 + p1;
#pragma unroll
            for (int off = 1; off < 16; off <<= 1)
                rsum += __shfl_xor(rsum, off, 32);
            li[r] = li[r] * alpha + rsum;
#pragma unroll
            for (int nt = 0; nt < 8; ++nt) o[nt][r] *= alpha;
            Psh[w][r + lhalf * 8][lmod]      = (_Float16)p0;
            Psh[w][r + lhalf * 8][16 + lmod] = (_Float16)p1;
        }

        // preload all V fragments + P, then the 8 independent PV WMMAs
        v16h vf[8];
#pragma unroll
        for (int nt = 0; nt < 8; ++nt)
            vf[nt] = fragB(&VshT[nt * 16 + lmod][0], lhalf);
        v16h pf = fragA(&Psh[w][lmod][0], lhalf);
#pragma unroll
        for (int nt = 0; nt < 8; ++nt)
            o[nt] = wmma_f16(pf, vf[nt], o[nt]);
        __syncthreads();
    }

    // normalize and write y_attn (b, t, h, hd) as (b*T+t)*C + h*HD
#pragma unroll
    for (int r = 0; r < 8; ++r) {
        const float inv = 1.0f / li[r];
        const int row = qbase + w * 16 + r + lhalf * 8;
        float* yp = Y + (size_t)(b * T_ + row) * C_ + h * HD_;
#pragma unroll
        for (int nt = 0; nt < 8; ++nt)
            yp[nt * 16 + lmod] = o[nt][r] * inv;
    }
}

// ---------------------------------------------------------------------------
extern "C" void kernel_launch(void* const* d_in, const int* in_sizes, int n_in,
                              void* d_out, int out_size, void* d_ws, size_t ws_size,
                              hipStream_t stream) {
    const float* x      = (const float*)d_in[0];
    const float* fcos   = (const float*)d_in[1];
    const float* fsin   = (const float*)d_in[2];
    const float* w_attn = (const float*)d_in[3];
    const float* b_attn = (const float*)d_in[4];
    const float* w_proj = (const float*)d_in[5];
    const float* b_proj = (const float*)d_in[6];

    float* out  = (float*)d_out;
    float* outK = out + (size_t)B_ * T_ * C_;                  // 8388608
    float* outV = outK + (size_t)B_ * NKV_ * T_ * HD_;         // +2097152

    float* qkv  = (float*)d_ws;                                // 4096*3072 f32
    float* yatt = qkv + (size_t)(B_ * T_) * QKVN_;             // 4096*2048 f32

    // 1) QKV = x @ w_attn^T + b_attn     (M=4096, N=3072, K=2048)
    gemm_bias_kernel<QKVN_, C_><<<dim3(QKVN_ / 128, (B_ * T_) / 64), 256, 0, stream>>>(
        x, w_attn, b_attn, qkv);

    // 2) RoPE Q (in place), RoPE K -> outK, copy V -> outV
    {
        const int tq = B_ * T_ * NH_ * (HD_ / 2);   // 4194304
        const int tk = B_ * T_ * NKV_ * (HD_ / 2);  // 1048576
        const int tv = B_ * T_ * NKV_ * HD_;        // 2097152
        rope_copy_kernel<<<tq / 256, 256, 0, stream>>>(qkv, outK, outV, fcos, fsin, 0, tq);
        rope_copy_kernel<<<tk / 256, 256, 0, stream>>>(qkv, outK, outV, fcos, fsin, 1, tk);
        rope_copy_kernel<<<tv / 256, 256, 0, stream>>>(qkv, outK, outV, fcos, fsin, 2, tv);
    }

    // 3) Flash attention -> yatt
    attn_kernel<<<dim3(T_ / 64, NH_, B_), 128, 0, stream>>>(qkv, outK, outV, yatt);

    // 4) y = yatt @ w_proj^T + b_proj    (M=4096, N=2048, K=2048)
    gemm_bias_kernel<C_, C_><<<dim3(C_ / 128, (B_ * T_) / 64), 256, 0, stream>>>(
        yatt, w_proj, b_proj, out);
}
